// RelationalHypergraphTransformer_90744069030528
// MI455X (gfx1250) — matrix operand order repaired and checked by reading, not compiled, not run on hardware
//
#include <hip/hip_runtime.h>
#include <math.h>

#define N_NODES 100000
#define N_EDGES 1600000
#define IN_DIM  128
#define HID     64
#define NH      4
#define DH      16
#define NLAYERS 3
#define OUT_DIM 64
#define SLOPE   0.2f
#define LN_EPS  1e-5f

typedef __attribute__((ext_vector_type(16))) __bf16 v16bf;
typedef __attribute__((ext_vector_type(8)))  float  v8f;

union BF16x16 {
    v16bf v;
    unsigned short u[16];
    uint4 q[2];
};

__device__ __forceinline__ unsigned short f32_to_bf16_rne(float f) {
    unsigned x = __float_as_uint(f);
    unsigned r = x + 0x7FFFu + ((x >> 16) & 1u);
    return (unsigned short)(r >> 16);
}

// order-preserving float<->uint encoding for atomicMax on f32
__device__ __forceinline__ unsigned ord_enc(float f) {
    unsigned u = __float_as_uint(f);
    return (u & 0x80000000u) ? ~u : (u | 0x80000000u);
}
__device__ __forceinline__ float ord_dec(unsigned u) {
    return (u & 0x80000000u) ? __uint_as_float(u & 0x7FFFFFFFu)
                             : __uint_as_float(~u);
}
__device__ __forceinline__ float lrelu(float x) { return x > 0.f ? x : SLOPE * x; }

// ---------------------------------------------------------------------------
// Pack f32 weight matrix (K x 64, row-major) into per-lane bf16 WMMA-B layout:
// for (kstep, nt, lane): 16 contiguous bf16 covering k = kstep*32 + (lane>>4)*8 + (j2&7) + (j2>>3)*16,
// n = (lane&15) + nt*16.  One 32-byte chunk per lane per fragment.
// ---------------------------------------------------------------------------
__global__ void pack_w_kernel(const float* __restrict__ W,
                              unsigned short* __restrict__ dst, int K) {
    int t = blockIdx.x * blockDim.x + threadIdx.x;
    if (t >= K * 64) return;
    int j2    = t & 15;
    int lane  = (t >> 4) & 31;
    int rest  = t >> 9;
    int nt    = rest & 3;
    int kstep = rest >> 2;
    int n = (lane & 15) + nt * 16;
    int k = kstep * 32 + (lane >> 4) * 8 + (j2 & 7) + (j2 >> 3) * 16;
    dst[t] = f32_to_bf16_rne(W[k * 64 + n]);
}

// ---------------------------------------------------------------------------
// Out[nrows x 64] = A[nrows x K] (f32) * Bpacked[K x 64] (bf16) + bias
// One wave per 16-row tile, 4 WMMA column tiles, fully unrolled K loop.
// Branch-free A loads: row index clamped (all tiles are full when nrows%16==0;
// stores are predicated for the general case).
// ---------------------------------------------------------------------------
template <int K>
__global__ void gemm_wmma_kernel(const float* __restrict__ A,
                                 const unsigned short* __restrict__ Bp,
                                 const float* __restrict__ bias,
                                 float* __restrict__ Out,
                                 int nrows) {
    const int lane = threadIdx.x & 31;
    const int wave = threadIdx.x >> 5;
    const int tile = blockIdx.x * (blockDim.x >> 5) + wave;
    const int numTiles = (nrows + 15) >> 4;
    if (tile >= numTiles) return;     // wave-uniform: EXEC stays all-ones for WMMA

    const int m    = lane & 15;
    const int half = lane >> 4;
    int row = tile * 16 + m;
    if (row >= nrows) row = nrows - 1;          // clamp: no divergent loads
    const float* Arow = A + (size_t)row * K;

    v8f acc[4];
#pragma unroll
    for (int i = 0; i < 4; ++i) acc[i] = (v8f)0.0f;

#pragma unroll
    for (int kk = 0; kk < K; kk += 32) {
        // A fragment (16-bit A 16x32 layout): per lane the 16 K-values are two
        // contiguous 8-float runs -> 4x float4 loads.
        const float* p0 = Arow + kk + half * 8;
        float4 r0 = *(const float4*)(p0);
        float4 r1 = *(const float4*)(p0 + 4);
        float4 r2 = *(const float4*)(p0 + 16);
        float4 r3 = *(const float4*)(p0 + 20);
        BF16x16 af;
        af.u[0]  = f32_to_bf16_rne(r0.x);  af.u[1]  = f32_to_bf16_rne(r0.y);
        af.u[2]  = f32_to_bf16_rne(r0.z);  af.u[3]  = f32_to_bf16_rne(r0.w);
        af.u[4]  = f32_to_bf16_rne(r1.x);  af.u[5]  = f32_to_bf16_rne(r1.y);
        af.u[6]  = f32_to_bf16_rne(r1.z);  af.u[7]  = f32_to_bf16_rne(r1.w);
        af.u[8]  = f32_to_bf16_rne(r2.x);  af.u[9]  = f32_to_bf16_rne(r2.y);
        af.u[10] = f32_to_bf16_rne(r2.z);  af.u[11] = f32_to_bf16_rne(r2.w);
        af.u[12] = f32_to_bf16_rne(r3.x);  af.u[13] = f32_to_bf16_rne(r3.y);
        af.u[14] = f32_to_bf16_rne(r3.z);  af.u[15] = f32_to_bf16_rne(r3.w);

        const int kstep = kk >> 5;
#pragma unroll
        for (int nt = 0; nt < 4; ++nt) {
            BF16x16 bf;
            const uint4* bp =
                (const uint4*)(Bp + (size_t)((kstep * 4 + nt) * 32 + lane) * 16);
            bf.q[0] = bp[0];
            bf.q[1] = bp[1];
            acc[nt] = __builtin_amdgcn_wmma_f32_16x16x32_bf16(
                false, af.v, false, bf.v, (short)0, acc[nt], false, false);
        }
    }

    // C/D layout: col = lane&15 (+nt*16), rows r + half*8 across 8 VGPRs
#pragma unroll
    for (int nt = 0; nt < 4; ++nt) {
        int col = (lane & 15) + nt * 16;
        float bv = bias ? bias[col] : 0.f;
#pragma unroll
        for (int r = 0; r < 8; ++r) {
            int orow = tile * 16 + r + half * 8;
            if (orow < nrows) Out[(size_t)orow * 64 + col] = acc[nt][r] + bv;
        }
    }
}

// el/er attention logits: thread per (node, head)
__global__ void elr_kernel(const float* __restrict__ z,
                           const float* __restrict__ al,
                           const float* __restrict__ ar,
                           float* __restrict__ el, float* __restrict__ er) {
    int t = blockIdx.x * blockDim.x + threadIdx.x;
    if (t >= N_NODES * NH) return;
    int n = t >> 2, h = t & 3;
    const float* zp  = z + (size_t)n * HID + h * DH;
    const float* alp = al + h * DH;
    const float* arp = ar + h * DH;
    float sl = 0.f, sr = 0.f;
#pragma unroll
    for (int d = 0; d < DH; ++d) { float v = zp[d]; sl += v * alp[d]; sr += v * arp[d]; }
    el[t] = sl;
    er[t] = sr;
}

__global__ void init_softmax_kernel(unsigned* __restrict__ m, float* __restrict__ den) {
    int t = blockIdx.x * blockDim.x + threadIdx.x;
    if (t >= N_NODES * NH) return;
    m[t]   = 0x007FFFFFu;   // ord_enc(-inf)
    den[t] = 0.f;
}

__global__ void edge_max_kernel(const int* __restrict__ src, const int* __restrict__ dst,
                                const float* __restrict__ el, const float* __restrict__ er,
                                unsigned* __restrict__ m) {
    int e = blockIdx.x * blockDim.x + threadIdx.x;
    if (e >= N_EDGES) return;
    int s = src[e], d = dst[e];
    float4 a = ((const float4*)el)[s];
    float4 b = ((const float4*)er)[d];
    unsigned* mp = m + (size_t)d * 4;
    atomicMax(&mp[0], ord_enc(lrelu(a.x + b.x)));
    atomicMax(&mp[1], ord_enc(lrelu(a.y + b.y)));
    atomicMax(&mp[2], ord_enc(lrelu(a.z + b.z)));
    atomicMax(&mp[3], ord_enc(lrelu(a.w + b.w)));
}

__global__ void edge_exp_kernel(const int* __restrict__ src, const int* __restrict__ dst,
                                const float* __restrict__ el, const float* __restrict__ er,
                                const unsigned* __restrict__ m,
                                float* __restrict__ ex, float* __restrict__ den) {
    int e = blockIdx.x * blockDim.x + threadIdx.x;
    if (e >= N_EDGES) return;
    int s = src[e], d = dst[e];
    float4 a = ((const float4*)el)[s];
    float4 b = ((const float4*)er)[d];
    uint4  mu = ((const uint4*)m)[d];
    float4 r;
    r.x = __expf(lrelu(a.x + b.x) - ord_dec(mu.x));
    r.y = __expf(lrelu(a.y + b.y) - ord_dec(mu.y));
    r.z = __expf(lrelu(a.z + b.z) - ord_dec(mu.z));
    r.w = __expf(lrelu(a.w + b.w) - ord_dec(mu.w));
    ((float4*)ex)[e] = r;
    float* dp = den + (size_t)d * 4;
    atomicAdd(&dp[0], r.x);
    atomicAdd(&dp[1], r.y);
    atomicAdd(&dp[2], r.z);
    atomicAdd(&dp[3], r.w);
}

// residual + GAT bias as the accumulator seed (out = h + b_gat)
__global__ void init_out_kernel(const float* __restrict__ h,
                                const float* __restrict__ bg,
                                float* __restrict__ out) {
    int t = blockIdx.x * blockDim.x + threadIdx.x;
    if (t >= N_NODES * HID) return;
    out[t] = h[t] + bg[t & 63];
}

// message scatter: thread per (edge, head), 16 f32 atomic adds each
__global__ void edge_msg_kernel(const int* __restrict__ src, const int* __restrict__ dst,
                                const float* __restrict__ z, const float* __restrict__ ex,
                                const float* __restrict__ den, float* __restrict__ out) {
    int t = blockIdx.x * blockDim.x + threadIdx.x;
    if (t >= N_EDGES * NH) return;
    int e = t >> 2, h = t & 3;
    int s = src[e], d = dst[e];
    const float* zp = z + (size_t)s * HID + h * DH;
    __builtin_prefetch(zp, 0, 0);
    float alpha = ex[t] / den[(size_t)d * 4 + h];
    float* op = out + (size_t)d * HID + h * DH;
#pragma unroll
    for (int d0 = 0; d0 < DH; d0 += 4) {
        float4 zv = *(const float4*)(zp + d0);
        atomicAdd(op + d0 + 0, zv.x * alpha);
        atomicAdd(op + d0 + 1, zv.y * alpha);
        atomicAdd(op + d0 + 2, zv.z * alpha);
        atomicAdd(op + d0 + 3, zv.w * alpha);
    }
}

__global__ void layernorm_kernel(const float* __restrict__ h,
                                 const float* __restrict__ g,
                                 const float* __restrict__ b,
                                 float* __restrict__ out) {
    int n = blockIdx.x * blockDim.x + threadIdx.x;
    if (n >= N_NODES) return;
    const float* hp = h + (size_t)n * HID;
    float s = 0.f, s2 = 0.f;
#pragma unroll
    for (int i = 0; i < HID; i += 4) {
        float4 v = *(const float4*)(hp + i);
        s  += v.x + v.y + v.z + v.w;
        s2 += v.x * v.x + v.y * v.y + v.z * v.z + v.w * v.w;
    }
    float mu  = s * (1.f / HID);
    float var = s2 * (1.f / HID) - mu * mu;
    float rs  = rsqrtf(var + LN_EPS);
    float* op = out + (size_t)n * HID;
    for (int i = 0; i < HID; ++i) op[i] = (hp[i] - mu) * rs * g[i] + b[i];
}

extern "C" void kernel_launch(void* const* d_in, const int* in_sizes, int n_in,
                              void* d_out, int out_size, void* d_ws, size_t ws_size,
                              hipStream_t stream) {
    (void)in_sizes; (void)n_in; (void)out_size; (void)ws_size;
    const float* X     = (const float*)d_in[0];
    const int*   src   = (const int*)d_in[1];
    const int*   dst   = (const int*)d_in[2];
    const float* W_in  = (const float*)d_in[3];
    const float* b_in  = (const float*)d_in[4];
    // d_in[5..8]: rewiring-MLP weights — edge_score is unused downstream, skipped
    const float* W_gat = (const float*)d_in[9];
    const float* a_l   = (const float*)d_in[10];
    const float* a_r   = (const float*)d_in[11];
    const float* b_gat = (const float*)d_in[12];
    const float* ln_g  = (const float*)d_in[13];
    const float* ln_b  = (const float*)d_in[14];
    const float* W_out = (const float*)d_in[15];
    const float* b_out = (const float*)d_in[16];

    char* ws = (char*)d_ws;
    size_t off = 0;
    auto alloc = [&](size_t bytes) -> char* {
        char* p = ws + off;
        off += (bytes + 255) & ~(size_t)255;
        return p;
    };
    float*    hA   = (float*)alloc((size_t)N_NODES * HID * 4);
    float*    hB   = (float*)alloc((size_t)N_NODES * HID * 4);
    float*    zb   = (float*)alloc((size_t)N_NODES * HID * 4);
    float*    el   = (float*)alloc((size_t)N_NODES * NH * 4);
    float*    er   = (float*)alloc((size_t)N_NODES * NH * 4);
    unsigned* mEnc = (unsigned*)alloc((size_t)N_NODES * NH * 4);
    float*    den  = (float*)alloc((size_t)N_NODES * NH * 4);
    float*    ex   = (float*)alloc((size_t)N_EDGES * NH * 4);
    unsigned short* pW_in  = (unsigned short*)alloc((size_t)IN_DIM * 64 * 2);
    unsigned short* pW_g0  = (unsigned short*)alloc((size_t)HID * 64 * 2);
    unsigned short* pW_g1  = (unsigned short*)alloc((size_t)HID * 64 * 2);
    unsigned short* pW_g2  = (unsigned short*)alloc((size_t)HID * 64 * 2);
    unsigned short* pW_out = (unsigned short*)alloc((size_t)HID * 64 * 2);
    unsigned short* pW_g[3] = {pW_g0, pW_g1, pW_g2};

    const int B = 256;
    // pack all weight matrices to bf16 WMMA-B layout
    pack_w_kernel<<<(IN_DIM * 64 + B - 1) / B, B, 0, stream>>>(W_in, pW_in, IN_DIM);
    for (int l = 0; l < NLAYERS; ++l)
        pack_w_kernel<<<(HID * 64 + B - 1) / B, B, 0, stream>>>(
            W_gat + (size_t)l * HID * 64, pW_g[l], HID);
    pack_w_kernel<<<(HID * 64 + B - 1) / B, B, 0, stream>>>(W_out, pW_out, HID);

    const int numTiles = (N_NODES + 15) / 16;
    const int gemmGrid = (numTiles + 7) / 8;   // 8 waves / block
    const int gNH  = (N_NODES * NH  + B - 1) / B;
    const int gNH64= (N_NODES * HID + B - 1) / B;
    const int gE   = (N_EDGES       + B - 1) / B;
    const int gEH  = (N_EDGES * NH  + B - 1) / B;

    // input projection: h = X @ W_in + b_in
    gemm_wmma_kernel<IN_DIM><<<gemmGrid, B, 0, stream>>>(X, pW_in, b_in, hA, N_NODES);

    for (int l = 0; l < NLAYERS; ++l) {
        float* cur = (l & 1) ? hB : hA;
        float* nxt = (l & 1) ? hA : hB;
        gemm_wmma_kernel<HID><<<gemmGrid, B, 0, stream>>>(cur, pW_g[l], nullptr, zb, N_NODES);
        elr_kernel<<<gNH, B, 0, stream>>>(zb, a_l + (size_t)l * 64, a_r + (size_t)l * 64, el, er);
        init_softmax_kernel<<<gNH, B, 0, stream>>>(mEnc, den);
        edge_max_kernel<<<gE, B, 0, stream>>>(src, dst, el, er, mEnc);
        edge_exp_kernel<<<gE, B, 0, stream>>>(src, dst, el, er, mEnc, ex, den);
        init_out_kernel<<<gNH64, B, 0, stream>>>(cur, b_gat + (size_t)l * 64, nxt);
        edge_msg_kernel<<<gEH, B, 0, stream>>>(src, dst, zb, ex, den, nxt);
    }

    // final h lives in hB after 3 layers (hA->hB->hA->hB)
    layernorm_kernel<<<(N_NODES + B - 1) / B, B, 0, stream>>>(hB, ln_g, ln_b, zb);
    gemm_wmma_kernel<HID><<<gemmGrid, B, 0, stream>>>(zb, pW_out, b_out, (float*)d_out, N_NODES);
}
